// WavetableSynth_67705864454784
// MI455X (gfx1250) — matrix operand order repaired and checked by reading, not compiled, not run on hardware
//
#include <hip/hip_runtime.h>
#include <math.h>

typedef _Float16 v16h __attribute__((ext_vector_type(16)));
typedef _Float16 v8h  __attribute__((ext_vector_type(8)));
typedef float    v8f  __attribute__((ext_vector_type(8)));

#define TABLE_SIZE    1024
#define N_TABLES      8
#define N_FRAMES      128
#define TOTAL_SAMPLES 32768
#define BATCH         2

#define MIN_FREQ       (40.0f   / 11025.0f)
#define MAX_FREQ       (2000.0f / 11025.0f)
#define FREQ_RANGE     (MAX_FREQ - MIN_FREQ)
#define MAX_FREQ_DELTA (10.0f   / 11025.0f)
#define SIGMA_INV      100.0f               /* 1/0.01 */
#define GNORM          39.894228040143274f  /* 1/(0.01*sqrt(2*pi)) */

// Linear-interp frame frequency at sample m (torch align_corners=False grid).
__device__ __forceinline__ float f_at(int m, const float* ff) {
    float c = ((float)m + 0.5f) * (1.0f / 256.0f) - 0.5f;
    c = fminf(fmaxf(c, 0.0f), 127.0f);
    int lo = (int)c;
    int hi = (lo + 1 < 127) ? lo + 1 : 127;
    float w = c - (float)lo;
    return ff[lo] + (ff[hi] - ff[lo]) * w;
}

// ---------------------------------------------------------------------------
// Prep: softmax+avg_pool7 of table selection (frame rate), per-table norms,
// and the per-sample phase cumsum (block scan, double accumulation).
// One block per batch, 256 threads.
// ---------------------------------------------------------------------------
__global__ __launch_bounds__(256) void wts_prep_kernel(
    const float* __restrict__ tsel,       // (B,8,128)
    const float* __restrict__ freq,       // (B,1,128)
    const float* __restrict__ init_freq,  // (B,1)
    const float* __restrict__ wavetables, // (8,1024)
    float* __restrict__ ws_phase,         // (B,32768) phase in [0,1)
    float* __restrict__ ws_ts,            // (B,8,128) pooled softmax
    float* __restrict__ ws_norm)          // (8) 1/(max|w|+1e-8)
{
    const int b = blockIdx.x;
    const int t = threadIdx.x;

    __shared__ float  sm[N_TABLES][N_FRAMES];
    __shared__ float  ff[N_FRAMES];
    __shared__ double part[256];

    // Per-table normalization reciprocals (block 0 only).
    if (b == 0 && t < N_TABLES) {
        float m = 0.0f;
        for (int s = 0; s < TABLE_SIZE; ++s)
            m = fmaxf(m, fabsf(wavetables[t * TABLE_SIZE + s]));
        ws_norm[t] = 1.0f / (m + 1e-8f);
    }

    // Softmax over tables per frame; frame-rate frequency.
    if (t < N_FRAMES) {
        float v[N_TABLES];
        float mx = -INFINITY;
        for (int k = 0; k < N_TABLES; ++k) {
            v[k] = tsel[(b * N_TABLES + k) * N_FRAMES + t];
            mx = fmaxf(mx, v[k]);
        }
        float s = 0.0f;
        for (int k = 0; k < N_TABLES; ++k) { v[k] = __expf(v[k] - mx); s += v[k]; }
        float inv = 1.0f / s;
        for (int k = 0; k < N_TABLES; ++k) sm[k][t] = v[k] * inv;

        float f0 = MIN_FREQ + FREQ_RANGE / (1.0f + __expf(-init_freq[b]));
        ff[t] = f0 + tanhf(freq[b * N_FRAMES + t]) * MAX_FREQ_DELTA;
    }
    __syncthreads();

    // avg_pool1d(k=7, pad=3, count_include_pad) at frame rate.
    for (int idx = t; idx < N_TABLES * N_FRAMES; idx += 256) {
        int k = idx / N_FRAMES, j = idx % N_FRAMES;
        float s = 0.0f;
        for (int d = -3; d <= 3; ++d) {
            int jj = j + d;
            if (jj >= 0 && jj < N_FRAMES) s += sm[k][jj];
        }
        ws_ts[(b * N_TABLES + k) * N_FRAMES + j] = s * (1.0f / 7.0f);
    }

    // Phase cumsum: 256 threads x 128 samples, scan partials in LDS.
    const int SPT = TOTAL_SAMPLES / 256; // 128
    const int m0 = t * SPT;
    double local = 0.0;
    for (int i = 0; i < SPT; ++i) local += (double)f_at(m0 + i, ff);
    part[t] = local;
    __syncthreads();
    if (t == 0) {
        double c = 0.0;
        for (int i = 0; i < 256; ++i) { double p = part[i]; part[i] = c; c += p; }
    }
    __syncthreads();
    double cum = part[t];
    for (int i = 0; i < SPT; ++i) {
        cum += (double)f_at(m0 + i, ff);
        ws_phase[b * TOTAL_SAMPLES + m0 + i] = (float)(cum - floor(cum));
    }
}

// One K=32 chunk: LDS A-fragment, register-generated Gaussian B-fragment,
// one v_wmma_f32_16x16x32_f16 into the given accumulator chain.
__device__ __forceinline__ v8f wts_wmma_chunk(const v8h* __restrict__ tabv,
                                              int rowBase, int half,
                                              int s0, float phase, v8f acc) {
    // A: row = col(table), K runs {half*8..+7} and {half*8+16..+23}, contiguous
    //    16-byte LDS vectors -> two ds_load_b128.
    const int h0 = rowBase + s0; // halves, 8-aligned
    v8h r1 = tabv[(h0 >> 3) + 0];
    v8h r2 = tabv[(h0 >> 3) + 2];
    v16h a = __builtin_shufflevector(r1, r2,
                                     0, 1, 2, 3, 4, 5, 6, 7,
                                     8, 9, 10, 11, 12, 13, 14, 15);

    // B: Gaussian read(s, n) for K = s0 + half*8 + (i&7) + (i>>3)*16, packed to f16.
    v16h bm;
    #pragma unroll
    for (int i = 0; i < 16; ++i) {
        float pos = (float)(s0 + half * 8 + (i & 7) + ((i >> 3) << 4)) * (1.0f / 1023.0f);
        float z = (pos - phase) * SIGMA_INV;
        bm[i] = (_Float16)(__expf(-0.5f * z * z) * GNORM);
    }

    return __builtin_amdgcn_wmma_f32_16x16x32_f16(
        /*neg_a=*/false, a, /*neg_b=*/false, bm,
        /*c_mod=*/(short)0, acc, /*reuse_a=*/false, /*reuse_b=*/false);
}

// ---------------------------------------------------------------------------
// Synth: g[k,n] = sum_s tables_f16[k,s] * gauss(s,n) via v_wmma_f32_16x16x32_f16.
// 8 waves/WG, 16 samples/wave -> 128 samples per WG; grid = B*T/128 = 512.
// Two independent WMMA accumulator chains (unroll x2) so exp/pack VALU for one
// chain co-executes with the other chain's WMMA and no WMMA->WMMA RAW stalls.
// ---------------------------------------------------------------------------
__global__ __launch_bounds__(256) void wts_synth_kernel(
    const float* __restrict__ env,        // (B,1,128)
    const float* __restrict__ wavetables, // (8,1024)
    const float* __restrict__ ws_phase,   // (B,32768)
    const float* __restrict__ ws_ts,      // (B,8,128)
    const float* __restrict__ ws_norm,    // (8)
    float* __restrict__ out)              // (B,1,32768)
{
    __shared__ _Float16 tabLDS[16 * TABLE_SIZE]; // 32 KB; rows 8..15 are zeros

    const int tid = threadIdx.x;

    // Stage max-normalized f16 tables (+ zero pad rows) into LDS.
    for (int idx = tid; idx < 16 * TABLE_SIZE; idx += 256) {
        int k = idx >> 10, s = idx & (TABLE_SIZE - 1);
        float v = (k < N_TABLES) ? wavetables[k * TABLE_SIZE + s] * ws_norm[k] : 0.0f;
        tabLDS[idx] = (_Float16)v;
    }
    __syncthreads();

    const int WGS_PER_BATCH = TOTAL_SAMPLES / 128; // 256
    const int b      = blockIdx.x / WGS_PER_BATCH;
    const int wgBase = (blockIdx.x % WGS_PER_BATCH) * 128;
    const int wave   = tid >> 5;
    const int lane   = tid & 31;
    const int half   = lane >> 4;   // which K sub-range this lane owns
    const int col    = lane & 15;   // WMMA row (A) / column (B,D)
    const int n      = wgBase + wave * 16 + col; // sample index for this lane

    const float phase   = ws_phase[b * TOTAL_SAMPLES + n];
    const int   rowBase = col * TABLE_SIZE + half * 8; // halves

    const v8h* tabv = (const v8h*)tabLDS;

    v8f acc0 = {}, acc1 = {};
    for (int c = 0; c < TABLE_SIZE / 32; c += 2) {
        acc0 = wts_wmma_chunk(tabv, rowBase, half, c * 32,      phase, acc0);
        acc1 = wts_wmma_chunk(tabv, rowBase, half, c * 32 + 32, phase, acc1);
    }
    v8f acc = acc0 + acc1;

    // Epilogue: lane<16 holds g[k][n] for k=0..7 in acc[0..7] (D layout).
    if (lane < 16) {
        float cc = ((float)n + 0.5f) * (1.0f / 256.0f) - 0.5f;
        cc = fminf(fmaxf(cc, 0.0f), 127.0f);
        int lo = (int)cc;
        int hi = (lo + 1 < 127) ? lo + 1 : 127;
        float w = cc - (float)lo;

        const float* tsb = ws_ts + b * N_TABLES * N_FRAMES;
        float dot = 0.0f;
        #pragma unroll
        for (int k = 0; k < N_TABLES; ++k) {
            float t0 = tsb[k * N_FRAMES + lo];
            float t1 = tsb[k * N_FRAMES + hi];
            dot += acc[k] * (t0 + (t1 - t0) * w);
        }
        float e0 = fabsf(env[b * N_FRAMES + lo]);
        float e1 = fabsf(env[b * N_FRAMES + hi]);
        out[b * TOTAL_SAMPLES + n] = dot * (e0 + (e1 - e0) * w);
    }
}

extern "C" void kernel_launch(void* const* d_in, const int* in_sizes, int n_in,
                              void* d_out, int out_size, void* d_ws, size_t ws_size,
                              hipStream_t stream) {
    (void)in_sizes; (void)n_in; (void)out_size; (void)ws_size;

    const float* env       = (const float*)d_in[0]; // (B,1,128)
    const float* tsel      = (const float*)d_in[1]; // (B,8,128)
    const float* freq      = (const float*)d_in[2]; // (B,1,128)
    /* d_in[3] = smoothness: dead in the reference (overwritten by 0.01) */
    const float* init_freq = (const float*)d_in[4]; // (B,1)
    const float* wav       = (const float*)d_in[5]; // (8,1024)

    float* ws       = (float*)d_ws;
    float* ws_phase = ws;                                   // B*T      = 65536 f
    float* ws_ts    = ws_phase + BATCH * TOTAL_SAMPLES;     // B*8*128  =  2048 f
    float* ws_norm  = ws_ts + BATCH * N_TABLES * N_FRAMES;  // 8 f

    wts_prep_kernel<<<BATCH, 256, 0, stream>>>(
        tsel, freq, init_freq, wav, ws_phase, ws_ts, ws_norm);

    wts_synth_kernel<<<BATCH * (TOTAL_SAMPLES / 128), 256, 0, stream>>>(
        env, wav, ws_phase, ws_ts, ws_norm, (float*)d_out);
}
